// Model_FE_Att_63419487093321
// MI455X (gfx1250) — matrix operand order repaired and checked by reading, not compiled, not run on hardware
//
#include <hip/hip_runtime.h>

// ---------------- problem constants ----------------
#define T_STEPS 64
#define B_ 128
#define H_ 128
#define L_ 512
#define S_ 512

// ---------------- vector types ----------------
typedef __bf16 v16bf __attribute__((ext_vector_type(16)));
typedef __bf16 bf8   __attribute__((ext_vector_type(8)));
typedef float  v8f   __attribute__((ext_vector_type(8)));
typedef float  f8    __attribute__((ext_vector_type(8)));

// ---------------- scalar helpers ----------------
__device__ __forceinline__ __bf16 f2bf(float f) {
    unsigned u = __float_as_uint(f);
    u += 0x7FFFu + ((u >> 16) & 1u);           // round-to-nearest-even
    unsigned short hv = (unsigned short)(u >> 16);
    __bf16 r; __builtin_memcpy(&r, &hv, 2); return r;
}
__device__ __forceinline__ float bf2f(__bf16 b) {
    unsigned short hv; __builtin_memcpy(&hv, &b, 2);
    return __uint_as_float(((unsigned)hv) << 16);
}
__device__ __forceinline__ float sigm(float x) { return 1.f / (1.f + expf(-x)); }

__device__ __forceinline__ v8f wmma_bf16(v16bf a, v16bf b, v8f c) {
    return __builtin_amdgcn_wmma_f32_16x16x32_bf16(false, a, false, b, (short)0, c,
                                                   false, false);
}

// ---------------- fragment loaders (wave32 CDNA5 layouts) ----------------
// A 16x32 bf16 fragment from row-major [*, lda]: lane m = row0+(lane&15),
// half=lane>>4; element runs {k0+half*8 .. +7} and {k0+16+half*8 .. +7}.
__device__ __forceinline__ v16bf load_a_bf16(const __bf16* A, int lda, int row0,
                                             int k0, int lane) {
    int m = row0 + (lane & 15);
    int half = lane >> 4;
    const __bf16* p = A + (size_t)m * lda + k0 + half * 8;
    bf8 lo = *(const bf8*)p;
    bf8 hi = *(const bf8*)(p + 16);
    return __builtin_shufflevector(lo, hi, 0,1,2,3,4,5,6,7,8,9,10,11,12,13,14,15);
}
// Same but converting from f32 source (for dynamic concat inputs).
__device__ __forceinline__ v16bf load_a_f32(const float* A, int lda, int row0,
                                            int k0, int lane) {
    int m = row0 + (lane & 15);
    int half = lane >> 4;
    const float* p = A + (size_t)m * lda + k0 + half * 8;
    f8 lo = *(const f8*)p;
    f8 hi = *(const f8*)(p + 16);
    v16bf f;
#pragma unroll
    for (int i = 0; i < 8; i++) { f[i] = f2bf(lo[i]); f[i + 8] = f2bf(hi[i]); }
    return f;
}
// B 32x16 fragment from PRE-TRANSPOSED bf16 BT [N][K]: lane n = n0+(lane&15),
// element runs {k0+half*8 .. +7} and {k0+16+half*8 .. +7} (matches A).
__device__ __forceinline__ v16bf load_b_bf16t(const __bf16* BT, int ldk, int n0,
                                              int k0, int lane) {
    int n = n0 + (lane & 15);
    int half = lane >> 4;
    const __bf16* p = BT + (size_t)n * ldk + k0 + half * 8;
    bf8 lo = *(const bf8*)p;
    bf8 hi = *(const bf8*)(p + 16);
    return __builtin_shufflevector(lo, hi, 0,1,2,3,4,5,6,7,8,9,10,11,12,13,14,15);
}

// ---------------- conversion / init kernels ----------------
__global__ void k_f2b(__bf16* __restrict__ dst, const float* __restrict__ src, int n) {
    for (int i = blockIdx.x * blockDim.x + threadIdx.x; i < n;
         i += gridDim.x * blockDim.x) dst[i] = f2bf(src[i]);
}
// src f32 [K][N] -> dst bf16 [N][K]
__global__ void k_transpose_f2b(__bf16* __restrict__ dst, const float* __restrict__ src,
                                int K, int N) {
    int id = blockIdx.x * blockDim.x + threadIdx.x;
    if (id < K * N) {
        int n = id / K, k = id % K;
        dst[id] = f2bf(src[(size_t)k * N + n]);
    }
}
__global__ void k_copy_f32(float* __restrict__ dst, const float* __restrict__ src, int n) {
    for (int i = blockIdx.x * blockDim.x + threadIdx.x; i < n;
         i += gridDim.x * blockDim.x) dst[i] = src[i];
}
__global__ void k_zero_f32(float* __restrict__ d, int n) {
    for (int i = blockIdx.x * blockDim.x + threadIdx.x; i < n;
         i += gridDim.x * blockDim.x) d[i] = 0.f;
}
__global__ void k_zero_b16(__bf16* __restrict__ d, int n) {
    for (int i = blockIdx.x * blockDim.x + threadIdx.x; i < n;
         i += gridDim.x * blockDim.x) d[i] = f2bf(0.f);
}

// ---------------- Wh_enc = states @ W_a  (65536x512x512, bf16 WMMA) ----------------
// 256 thr = 8 waves; workgroup tile 128(M) x 64(N); wave = 16x64 (4 accums).
__global__ void k_whenc(__bf16* __restrict__ whenc, const __bf16* __restrict__ A,
                        const __bf16* __restrict__ BT) {
    int lane = threadIdx.x & 31, wave = threadIdx.x >> 5;
    int bn = blockIdx.x & 7;        // 8 N-blocks of 64
    int bm = blockIdx.x >> 3;       // 512 M-blocks of 128
    int row0 = bm * 128 + wave * 16;
    int n0 = bn * 64;
    v8f acc[4] = {};
    for (int k0 = 0; k0 < S_; k0 += 32) {
        v16bf a = load_a_bf16(A, S_, row0, k0, lane);
#pragma unroll
        for (int j = 0; j < 4; j++) {
            v16bf b = load_b_bf16t(BT, S_, n0 + j * 16, k0, lane);
            acc[j] = wmma_bf16(a, b, acc[j]);
        }
    }
    int nl = lane & 15, half = lane >> 4;
#pragma unroll
    for (int j = 0; j < 4; j++)
#pragma unroll
        for (int i = 0; i < 8; i++)
            whenc[(size_t)(row0 + i + half * 8) * S_ + n0 + j * 16 + nl] = f2bf(acc[j][i]);
}

// ---------------- x = concat(inp, attn) @ W_in + b_in  -> bf16 [B][H] ----------------
__global__ void k_xin(__bf16* __restrict__ xbf, const float* __restrict__ inp,
                      const float* __restrict__ attn, const __bf16* __restrict__ WT,
                      const float* __restrict__ bias) {
    int lane = threadIdx.x & 31, wave = threadIdx.x >> 5;
    int tile = blockIdx.x * 4 + wave;       // 64 tiles: 8(M) x 8(N)
    int row0 = (tile & 7) * 16, n0 = (tile >> 3) * 16;
    v8f acc = {};
    for (int k0 = 0; k0 < H_ + S_; k0 += 32) {
        v16bf a = (k0 < H_) ? load_a_f32(inp, H_, row0, k0, lane)
                            : load_a_f32(attn, S_, row0, k0 - H_, lane);
        v16bf b = load_b_bf16t(WT, H_ + S_, n0, k0, lane);
        acc = wmma_bf16(a, b, acc);
    }
    int nl = lane & 15, half = lane >> 4;
    float bv = bias[n0 + nl];
#pragma unroll
    for (int i = 0; i < 8; i++)
        xbf[(size_t)(row0 + i + half * 8) * H_ + n0 + nl] = f2bf(acc[i] + bv);
}

// ------- gates = x@Wx + h@Wh + b ; fused LSTM elementwise (wave owns i/f/g/o) -------
__global__ void k_gates(float* __restrict__ c, float* __restrict__ hF_new,
                        __bf16* __restrict__ hB_new, const __bf16* __restrict__ xbf,
                        const __bf16* __restrict__ hB_old,
                        const __bf16* __restrict__ WxT, const __bf16* __restrict__ WhT,
                        const float* __restrict__ bcell) {
    int lane = threadIdx.x & 31, wave = threadIdx.x >> 5;
    int tile = blockIdx.x * 4 + wave;       // 64 tiles of (b,j) in [128][128]
    int row0 = (tile & 7) * 16, j0 = (tile >> 3) * 16;
    v8f acc[4] = {};                        // i, f, g, o
    for (int k0 = 0; k0 < H_; k0 += 32) {
        v16bf ax = load_a_bf16(xbf, H_, row0, k0, lane);
#pragma unroll
        for (int g = 0; g < 4; g++)
            acc[g] = wmma_bf16(ax, load_b_bf16t(WxT, H_, g * H_ + j0, k0, lane), acc[g]);
        v16bf ah = load_a_bf16(hB_old, H_, row0, k0, lane);
#pragma unroll
        for (int g = 0; g < 4; g++)
            acc[g] = wmma_bf16(ah, load_b_bf16t(WhT, H_, g * H_ + j0, k0, lane), acc[g]);
    }
    int nl = lane & 15, half = lane >> 4, col = j0 + nl;
    float bi = bcell[col], bff = bcell[col + H_], bg = bcell[col + 2 * H_],
          bo = bcell[col + 3 * H_];
#pragma unroll
    for (int i = 0; i < 8; i++) {
        int idx = (row0 + i + half * 8) * H_ + col;
        float iv = sigm(acc[0][i] + bi);
        float fv = sigm(acc[1][i] + bff);
        float gv = tanhf(acc[2][i] + bg);
        float ov = sigm(acc[3][i] + bo);
        float cn = fv * c[idx] + iv * gv;
        float hn = ov * tanhf(cn);
        c[idx] = cn;
        hF_new[idx] = hn;
        hB_new[idx] = f2bf(hn);
    }
}

// ---------------- y = h @ U_a  (128x512, K=128) ----------------
__global__ void k_y(float* __restrict__ y, const __bf16* __restrict__ hB,
                    const __bf16* __restrict__ UaT) {
    int lane = threadIdx.x & 31, wave = threadIdx.x >> 5;
    int tile = blockIdx.x * 4 + wave;       // 256 tiles: 8(M) x 32(N)
    int row0 = (tile & 7) * 16, n0 = (tile >> 3) * 16;
    v8f acc = {};
    for (int k0 = 0; k0 < H_; k0 += 32) {
        v16bf a = load_a_bf16(hB, H_, row0, k0, lane);
        v16bf b = load_b_bf16t(UaT, H_, n0, k0, lane);
        acc = wmma_bf16(a, b, acc);
    }
    int nl = lane & 15, half = lane >> 4;
#pragma unroll
    for (int i = 0; i < 8; i++)
        y[(size_t)(row0 + i + half * 8) * S_ + n0 + nl] = acc[i];
}

// ---- s = v . tanh(Wh_enc + y), softmax over L; one workgroup per batch row ----
__global__ void __launch_bounds__(512) k_scores(float* __restrict__ a_out,
                                                float* __restrict__ hist,
                                                const __bf16* __restrict__ whenc,
                                                const float* __restrict__ y,
                                                const float* __restrict__ vatt) {
    __shared__ float sy[S_], sv[S_], ss[L_], red[512];
    int b = blockIdx.x, tid = threadIdx.x;
    sy[tid] = y[(size_t)b * S_ + tid];
    sv[tid] = vatt[tid];
    __syncthreads();
    int lane = tid & 31, wave = tid >> 5;
    for (int l = wave; l < L_; l += 16) {
        const __bf16* row = whenc + ((size_t)b * L_ + l) * S_ + lane * 16;
        __builtin_prefetch(row + 16 * S_, 0, 1);   // next row for this wave
        v16bf rv = *(const v16bf*)row;
        float acc = 0.f;
#pragma unroll
        for (int i = 0; i < 16; i++) {
            int v = lane * 16 + i;
            acc += sv[v] * tanhf(bf2f(rv[i]) + sy[v]);
        }
#pragma unroll
        for (int off = 16; off > 0; off >>= 1) acc += __shfl_xor(acc, off, 32);
        if (lane == 0) ss[l] = acc;
    }
    __syncthreads();
    float s = ss[tid];
    red[tid] = s; __syncthreads();
    for (int st = 256; st > 0; st >>= 1) {
        if (tid < st) red[tid] = fmaxf(red[tid], red[tid + st]);
        __syncthreads();
    }
    float mx = red[0]; __syncthreads();
    float e = expf(s - mx);
    red[tid] = e; __syncthreads();
    for (int st = 256; st > 0; st >>= 1) {
        if (tid < st) red[tid] += red[tid + st];
        __syncthreads();
    }
    float av = e / red[0];
    a_out[(size_t)b * L_ + tid] = av;
    hist[(size_t)b * L_ + tid] = av;
}

// ---------------- d[b,s] = sum_l a[b,l] * states[b,l,s]  ----------------
__global__ void k_ctx(float* __restrict__ dF, __bf16* __restrict__ dB,
                      const float* __restrict__ a, const __bf16* __restrict__ states) {
    __shared__ float sa[L_];
    int b = blockIdx.x, tid = threadIdx.x;     // 256 threads, 2 cols each
    sa[tid] = a[(size_t)b * L_ + tid];
    sa[tid + 256] = a[(size_t)b * L_ + tid + 256];
    __syncthreads();
    int s0 = tid * 2;
    const __bf16* base = states + (size_t)b * L_ * S_ + s0;
    float acc0 = 0.f, acc1 = 0.f;
    for (int l = 0; l < L_; l++) {
        const __bf16* p = base + (size_t)l * S_;
        float al = sa[l];
        acc0 += al * bf2f(p[0]);
        acc1 += al * bf2f(p[1]);
    }
    dF[(size_t)b * S_ + s0] = acc0;
    dF[(size_t)b * S_ + s0 + 1] = acc1;
    dB[(size_t)b * S_ + s0] = f2bf(acc0);
    dB[(size_t)b * S_ + s0 + 1] = f2bf(acc1);
}

// ---------------- out = concat(h, d) @ W_out + b_out  -> f32 [B][H] ----------------
__global__ void k_out(float* __restrict__ out, const __bf16* __restrict__ hB,
                      const __bf16* __restrict__ dB, const __bf16* __restrict__ WT,
                      const float* __restrict__ bias) {
    int lane = threadIdx.x & 31, wave = threadIdx.x >> 5;
    int tile = blockIdx.x * 4 + wave;
    int row0 = (tile & 7) * 16, n0 = (tile >> 3) * 16;
    v8f acc = {};
    for (int k0 = 0; k0 < H_ + S_; k0 += 32) {
        v16bf a = (k0 < H_) ? load_a_bf16(hB, H_, row0, k0, lane)
                            : load_a_bf16(dB, S_, row0, k0 - H_, lane);
        v16bf b = load_b_bf16t(WT, H_ + S_, n0, k0, lane);
        acc = wmma_bf16(a, b, acc);
    }
    int nl = lane & 15, half = lane >> 4;
    float bv = bias[n0 + nl];
#pragma unroll
    for (int i = 0; i < 8; i++)
        out[(size_t)(row0 + i + half * 8) * H_ + n0 + nl] = acc[i] + bv;
}

// ---------------- host ----------------
extern "C" void kernel_launch(void* const* d_in, const int* in_sizes, int n_in,
                              void* d_out, int out_size, void* d_ws, size_t ws_size,
                              hipStream_t stream) {
    (void)in_sizes; (void)n_in; (void)out_size; (void)ws_size;
    const float* dec    = (const float*)d_in[0];
    const float* states = (const float*)d_in[1];
    const float* c0     = (const float*)d_in[2];
    const float* h0     = (const float*)d_in[3];
    const float* W_in   = (const float*)d_in[4];
    const float* b_in   = (const float*)d_in[5];
    const float* Wx     = (const float*)d_in[6];
    const float* Wh     = (const float*)d_in[7];
    const float* b_cell = (const float*)d_in[8];
    const float* W_a    = (const float*)d_in[9];
    const float* U_a    = (const float*)d_in[10];
    const float* v_att  = (const float*)d_in[11];
    const float* W_out  = (const float*)d_in[12];
    const float* b_out  = (const float*)d_in[13];

    float* outp = (float*)d_out;                 // outputs [T,B,H] at 0
    float* h_final = outp + (size_t)T_STEPS * B_ * H_;          // [B,H]
    float* hist = h_final + (size_t)B_ * H_;     // attn_hist [T,B,L]

    char* cur = (char*)d_ws;
    auto take = [&](size_t bytes) -> char* {
        char* p = cur; cur += (bytes + 255) & ~(size_t)255; return p;
    };
    const size_t NBL = (size_t)B_ * L_ * S_;
    __bf16* states_bf = (__bf16*)take(NBL * 2);
    __bf16* whenc_bf  = (__bf16*)take(NBL * 2);
    __bf16* WaT   = (__bf16*)take((size_t)S_ * S_ * 2);
    __bf16* WinT  = (__bf16*)take((size_t)H_ * (H_ + S_) * 2);
    __bf16* WxT   = (__bf16*)take((size_t)4 * H_ * H_ * 2);
    __bf16* WhT   = (__bf16*)take((size_t)4 * H_ * H_ * 2);
    __bf16* UaT   = (__bf16*)take((size_t)S_ * H_ * 2);
    __bf16* WoutT = (__bf16*)take((size_t)H_ * (H_ + S_) * 2);
    __bf16* x_bf  = (__bf16*)take((size_t)B_ * H_ * 2);
    float*  h_f[2]; h_f[0] = (float*)take((size_t)B_ * H_ * 4);
                    h_f[1] = (float*)take((size_t)B_ * H_ * 4);
    __bf16* h_b[2]; h_b[0] = (__bf16*)take((size_t)B_ * H_ * 2);
                    h_b[1] = (__bf16*)take((size_t)B_ * H_ * 2);
    float*  c_ws  = (float*)take((size_t)B_ * H_ * 4);
    float*  y_ws  = (float*)take((size_t)B_ * S_ * 4);
    float*  a_ws  = (float*)take((size_t)B_ * L_ * 4);
    float*  d_f   = (float*)take((size_t)B_ * S_ * 4);
    __bf16* d_b   = (__bf16*)take((size_t)B_ * S_ * 2);

    // --- one-time conversions / init ---
    k_f2b<<<8192, 256, 0, stream>>>(states_bf, states, (int)NBL);
    k_transpose_f2b<<<(S_ * S_ + 255) / 256, 256, 0, stream>>>(WaT, W_a, S_, S_);
    k_transpose_f2b<<<((H_ + S_) * H_ + 255) / 256, 256, 0, stream>>>(WinT, W_in, H_ + S_, H_);
    k_transpose_f2b<<<(H_ * 4 * H_ + 255) / 256, 256, 0, stream>>>(WxT, Wx, H_, 4 * H_);
    k_transpose_f2b<<<(H_ * 4 * H_ + 255) / 256, 256, 0, stream>>>(WhT, Wh, H_, 4 * H_);
    k_transpose_f2b<<<(H_ * S_ + 255) / 256, 256, 0, stream>>>(UaT, U_a, H_, S_);
    k_transpose_f2b<<<((H_ + S_) * H_ + 255) / 256, 256, 0, stream>>>(WoutT, W_out, H_ + S_, H_);
    k_copy_f32<<<64, 256, 0, stream>>>(c_ws, c0, B_ * H_);
    k_copy_f32<<<64, 256, 0, stream>>>(h_f[0], h0, B_ * H_);
    k_f2b<<<64, 256, 0, stream>>>(h_b[0], h0, B_ * H_);
    k_zero_f32<<<256, 256, 0, stream>>>(d_f, B_ * S_);
    k_zero_b16<<<256, 256, 0, stream>>>(d_b, B_ * S_);

    // --- Wh_enc precompute ---
    k_whenc<<<4096, 256, 0, stream>>>(whenc_bf, states_bf, WaT);

    // --- recurrence ---
    for (int t = 0; t < T_STEPS; t++) {
        int rd = t & 1, wr = (t + 1) & 1;
        k_xin<<<16, 128, 0, stream>>>(x_bf, dec + (size_t)t * B_ * H_, d_f, WinT, b_in);
        k_gates<<<16, 128, 0, stream>>>(c_ws, h_f[wr], h_b[wr], x_bf, h_b[rd],
                                        WxT, WhT, b_cell);
        k_y<<<64, 128, 0, stream>>>(y_ws, h_b[wr], UaT);
        k_scores<<<B_, 512, 0, stream>>>(a_ws, hist + (size_t)t * B_ * L_,
                                         whenc_bf, y_ws, v_att);
        k_ctx<<<B_, 256, 0, stream>>>(d_f, d_b, a_ws, states_bf);
        k_out<<<16, 128, 0, stream>>>(outp + (size_t)t * B_ * H_, h_b[wr], d_b,
                                      WoutT, b_out);
    }
    // after step T-1, newest h lives in index (T_STEPS) & 1 == 0
    k_copy_f32<<<64, 256, 0, stream>>>(h_final, h_f[0], B_ * H_);
}
